// GIPAConv_52243982189091
// MI455X (gfx1250) — matrix operand order repaired and checked by reading, not compiled, not run on hardware
//
#include <hip/hip_runtime.h>

// ---------------------------------------------------------------------------
// GIPAConv for MI455X (gfx1250, wave32)
//   fs/fd projection: v_wmma_f32_16x16x32_f16 (f16 in, f32 accumulate)
//   edge aggregation: global_atomic_add_f32 (hardware fp32 atomic, L2-resident)
// ---------------------------------------------------------------------------

#define N_NODES   50000
#define N_EDGES   800000
#define IN_FEATS  128
#define EDGE_FEATS 8
#define N_HEADS   2
#define OUT_FEATS 32
#define HF        64          // N_HEADS * OUT_FEATS
#define NEG_SLOPE 0.2f

typedef __attribute__((ext_vector_type(16))) _Float16 v16h;
typedef __attribute__((ext_vector_type(8)))  float    v8f;
typedef __attribute__((ext_vector_type(4)))  float    v4f;

// -------------------------------------------------------------------------
// Kernel 1: pack [W_src | W_dst] (f32, [128,64] each) into f16 B-fragments,
// pre-swizzled into WMMA 32x16 B layout so the GEMM does one contiguous
// 32-byte load per lane per K-step.
// Layout: W_frag[((ct*4 + ks)*32 + lane)*16 + e]
//   col  = ct*16 + (lane&15)
//   K    = ks*32 + ((lane&16)?8:0) + (e&7) + ((e&8)?16:0)   (e = 0..15, 2 per VGPR)
// -------------------------------------------------------------------------
__global__ void prep_wfrag_kernel(const float* __restrict__ W_src,
                                  const float* __restrict__ W_dst,
                                  _Float16* __restrict__ W_frag) {
    int t = blockIdx.x * blockDim.x + threadIdx.x;      // 0 .. 16383
    if (t >= 8 * 4 * 32 * 16) return;
    int e    = t & 15;
    int lane = (t >> 4) & 31;
    int ks   = (t >> 9) & 3;
    int ct   = t >> 11;
    int col  = ct * 16 + (lane & 15);
    int k    = ks * 32 + ((lane & 16) ? 8 : 0) + (e & 7) + ((e & 8) ? 16 : 0);
    float v  = (col < HF) ? W_src[k * HF + col] : W_dst[k * HF + (col - HF)];
    W_frag[t] = (_Float16)v;
}

// -------------------------------------------------------------------------
// Kernel 2: projection GEMM  proj[50000,128] = feat[50000,128] @ [W_src|W_dst]
// One block = 16 node rows; 8 waves = 8 column tiles of 16.
// cols 0..63  -> fs  (workspace, gathered by edge kernel)
// cols 64..127-> fd + b_dst -> written straight to d_out (residual term)
// -------------------------------------------------------------------------
__global__ void __launch_bounds__(256)
gemm_kernel(const float* __restrict__ feat,
            const _Float16* __restrict__ W_frag,
            const float* __restrict__ b_dst,
            float* __restrict__ fs_ws,
            float* __restrict__ out) {
    const int lane = threadIdx.x & 31;
    const int wv   = threadIdx.x >> 5;            // column tile 0..7
    const int row_base = blockIdx.x * 16;         // 50000 = 16 * 3125 exact
    const int r     = row_base + (lane & 15);
    const int khalf = (lane & 16) ? 8 : 0;        // A-fragment K sub-half

    v8f c = {0.f, 0.f, 0.f, 0.f, 0.f, 0.f, 0.f, 0.f};

    const v16h* bfrag = (const v16h*)W_frag + (wv * 4) * 32 + lane;

#pragma unroll
    for (int ks = 0; ks < 4; ++ks) {
        const int bk = ks * 32;
        // --- A fragment: 16x32 f16, ISA layout (lane<16: K 0-7,16-23; lane>=16: K 8-15,24-31)
        const float* rowp = feat + (size_t)r * IN_FEATS + bk + khalf;
        v4f q0 = *(const v4f*)(rowp + 0);
        v4f q1 = *(const v4f*)(rowp + 4);
        v4f q2 = *(const v4f*)(rowp + 16);
        v4f q3 = *(const v4f*)(rowp + 20);
        v16h a;
#pragma unroll
        for (int i = 0; i < 4; ++i) {
            a[i]      = (_Float16)q0[i];
            a[4 + i]  = (_Float16)q1[i];
            a[8 + i]  = (_Float16)q2[i];
            a[12 + i] = (_Float16)q3[i];
        }
        // --- B fragment: pre-swizzled, one 32B load
        v16h b = bfrag[ks * 32];

        c = __builtin_amdgcn_wmma_f32_16x16x32_f16(
                /*neg_a=*/false, a, /*neg_b=*/false, b,
                /*c_mod=*/(short)0, c, /*reuse_a=*/false, /*reuse_b=*/false);
    }

    // --- epilogue: C layout => lane<16: M=i, N=lane; lane>=16: M=8+i, N=lane-16
    const int ncol  = wv * 16 + (lane & 15);
    const int mbase = row_base + ((lane & 16) ? 8 : 0);
    if (wv < 4) {                                    // fs columns
#pragma unroll
        for (int i = 0; i < 8; ++i)
            fs_ws[(size_t)(mbase + i) * HF + ncol] = c[i];
    } else {                                         // fd columns (+bias) -> output
        const int oc = ncol - HF;
        const float bb = b_dst[oc];
#pragma unroll
        for (int i = 0; i < 8; ++i)
            out[(size_t)(mbase + i) * HF + oc] = c[i] + bb;
    }
}

// -------------------------------------------------------------------------
// Kernel 3: attention scalars  a_src/a_dst[N,2] = feat @ W_attn_{src,dst}
// wave-per-node, coalesced feature reads, __shfl_xor butterfly (wave32).
// -------------------------------------------------------------------------
__global__ void __launch_bounds__(256)
attn_kernel(const float* __restrict__ feat,
            const float* __restrict__ Wa_s,
            const float* __restrict__ Wa_d,
            float* __restrict__ a_src,
            float* __restrict__ a_dst) {
    const int gid  = blockIdx.x * blockDim.x + threadIdx.x;
    const int node = gid >> 5;
    const int lane = gid & 31;
    if (node >= N_NODES) return;

    const float* fr = feat + (size_t)node * IN_FEATS;
    float s0 = 0.f, s1 = 0.f, d0 = 0.f, d1 = 0.f;
#pragma unroll
    for (int t = 0; t < 4; ++t) {
        const int k = lane + 32 * t;
        const float f = fr[k];
        s0 = fmaf(f, Wa_s[k * 2 + 0], s0);
        s1 = fmaf(f, Wa_s[k * 2 + 1], s1);
        d0 = fmaf(f, Wa_d[k * 2 + 0], d0);
        d1 = fmaf(f, Wa_d[k * 2 + 1], d1);
    }
#pragma unroll
    for (int off = 16; off > 0; off >>= 1) {
        s0 += __shfl_xor(s0, off, 32);
        s1 += __shfl_xor(s1, off, 32);
        d0 += __shfl_xor(d0, off, 32);
        d1 += __shfl_xor(d1, off, 32);
    }
    if (lane == 0) {
        a_src[node * 2 + 0] = s0;
        a_src[node * 2 + 1] = s1;
        a_dst[node * 2 + 0] = d0;
        a_dst[node * 2 + 1] = d1;
    }
}

// hardware fp32 scatter-add (no CAS fallback); fire-and-forget, implicit
// wait-idle at s_endpgm covers outstanding STOREcnt.
__device__ __forceinline__ void gadd_f32(float* p, float v) {
    asm volatile("global_atomic_add_f32 %0, %1, off" :: "v"(p), "v"(v) : "memory");
}

// -------------------------------------------------------------------------
// Kernel 4: edge phase. wave-per-edge: broadcast attention math, lane owns
// message floats [2*lane, 2*lane+1] (same head per lane since 2*lane even).
// -------------------------------------------------------------------------
__global__ void __launch_bounds__(256)
edge_kernel(const float* __restrict__ fs_ws,
            const float* __restrict__ a_src,
            const float* __restrict__ a_dst,
            const float* __restrict__ feat_edge,
            const float* __restrict__ W_ae,
            const int* __restrict__ src,
            const int* __restrict__ dst,
            float* __restrict__ out) {
    const int lane = threadIdx.x & 31;
    const int e    = blockIdx.x * (blockDim.x >> 5) + (threadIdx.x >> 5);
    const int s = src[e];
    const int d = dst[e];

    // a_edge (broadcast across wave; W_ae is 64B, cache-resident)
    float ae0 = 0.f, ae1 = 0.f;
    const float* fe = feat_edge + (size_t)e * EDGE_FEATS;
#pragma unroll
    for (int j = 0; j < EDGE_FEATS; ++j) {
        const float f = fe[j];
        ae0 = fmaf(f, W_ae[j * 2 + 0], ae0);
        ae1 = fmaf(f, W_ae[j * 2 + 1], ae1);
    }
    float e0 = a_src[s * 2 + 0] + a_dst[d * 2 + 0] + ae0;
    float e1 = a_src[s * 2 + 1] + a_dst[d * 2 + 1] + ae1;
    e0 = (e0 > 0.f) ? e0 : NEG_SLOPE * e0;
    e1 = (e1 > 0.f) ? e1 : NEG_SLOPE * e1;

    const int j0 = lane * 2;
    const float w = (j0 < OUT_FEATS) ? e0 : e1;   // head 0 : head 1
    const float2 f2 = *(const float2*)(fs_ws + (size_t)s * HF + j0);
    float* op = out + (size_t)d * HF + j0;
    gadd_f32(op + 0, f2.x * w);
    gadd_f32(op + 1, f2.y * w);
}

// ---------------------------------------------------------------------------
extern "C" void kernel_launch(void* const* d_in, const int* in_sizes, int n_in,
                              void* d_out, int out_size, void* d_ws, size_t ws_size,
                              hipStream_t stream) {
    const float* feat_src    = (const float*)d_in[0];
    const float* feat_edge   = (const float*)d_in[1];
    const int*   src         = (const int*)d_in[2];
    const int*   dst         = (const int*)d_in[3];
    const float* W_src       = (const float*)d_in[4];
    const float* W_dst       = (const float*)d_in[5];
    const float* b_dst       = (const float*)d_in[6];
    const float* W_attn_src  = (const float*)d_in[7];
    const float* W_attn_dst  = (const float*)d_in[8];
    const float* W_attn_edge = (const float*)d_in[9];
    float* out = (float*)d_out;

    // workspace layout (~13.6 MB): W_frag f16 | fs | a_src | a_dst
    char* ws = (char*)d_ws;
    _Float16* W_frag = (_Float16*)ws;                             // 32768 B
    float*    fs_ws  = (float*)(ws + 32768);                      // 12.8 MB
    float*    a_srcp = (float*)(ws + 32768 + (size_t)N_NODES * HF * 4);
    float*    a_dstp = a_srcp + (size_t)N_NODES * N_HEADS;

    // 1) pack weights into WMMA B-fragment order (f16)
    prep_wfrag_kernel<<<64, 256, 0, stream>>>(W_src, W_dst, W_frag);
    // 2) projection GEMM: fs -> ws, fd+b -> out (fully overwrites d_out)
    gemm_kernel<<<N_NODES / 16, 256, 0, stream>>>(feat_src, W_frag, b_dst, fs_ws, out);
    // 3) attention scalars (wave-per-node)
    attn_kernel<<<(N_NODES * 32) / 256, 256, 0, stream>>>(feat_src, W_attn_src,
                                                          W_attn_dst, a_srcp, a_dstp);
    // 4) edge gather / weight / scatter-add (wave-per-edge)
    edge_kernel<<<(N_EDGES * 32) / 256, 256, 0, stream>>>(fs_ws, a_srcp, a_dstp,
                                                          feat_edge, W_attn_edge,
                                                          src, dst, out);
}
